// DynamicConvBlock_3719441679005
// MI455X (gfx1250) — matrix operand order repaired and checked by reading, not compiled, not run on hardware
//
#include <hip/hip_runtime.h>
#include <hip/hip_bf16.h>
#include <math.h>

// ---------------- problem constants ----------------
#define BATCH 32
#define CIN   128
#define COUT  256
#define KSZ0  16          // K
#define LIN   2048        // L
#define HID   32
#define MAXK  24          // round(1.5*K)
#define PADW  12          // MAXK/2
#define LOUT  (LIN + 2*PADW - MAXK + 1)   // 2049
#define KTOT  (CIN * MAXK)                // 3072
#define BN_EPS 1e-5f

typedef __attribute__((ext_vector_type(16))) __bf16 v16bf;
typedef __attribute__((ext_vector_type(8)))  __bf16 v8bf;
typedef __attribute__((ext_vector_type(8)))  float  v8f;
typedef __attribute__((ext_vector_type(4)))  unsigned int uint32x4;
typedef __attribute__((ext_vector_type(8)))  int  int32x8;
typedef __attribute__((ext_vector_type(4)))  int  int32x4;

#if __has_builtin(__builtin_amdgcn_tensor_load_to_lds)
#define USE_TDM 1
#else
#define USE_TDM 0
#endif

static __device__ __forceinline__ void wait_tensorcnt0() {
#if __has_builtin(__builtin_amdgcn_s_wait_tensorcnt)
    __builtin_amdgcn_s_wait_tensorcnt(0);
#else
    asm volatile("s_wait_tensorcnt 0x0" ::: "memory");
#endif
}

// =====================================================================
// Kernel 1: predictor network -> per-batch interpolation tables
// =====================================================================
__global__ __launch_bounds__(256)
void dcb_predict_kernel(const float* __restrict__ x,
                        const float* __restrict__ pred_w,
                        const float* __restrict__ pred_b,
                        const float* __restrict__ bn_gamma,
                        const float* __restrict__ bn_beta,
                        const float* __restrict__ bn_mean,
                        const float* __restrict__ bn_var,
                        const float* __restrict__ lin_w,
                        const float* __restrict__ lin_b,
                        int*   __restrict__ i0_out,
                        int*   __restrict__ i1_out,
                        float* __restrict__ w_out)
{
    __shared__ float pw[HID * CIN];           // 16 KB
    __shared__ float sc[HID], bi[HID], pb[HID];
    __shared__ float red[256 * HID];          // 32 KB

    const int b   = blockIdx.x;
    const int tid = threadIdx.x;

    for (int e = tid; e < HID * CIN; e += 256) pw[e] = pred_w[e];
    if (tid < HID) {
        float inv = rsqrtf(bn_var[tid] + BN_EPS);
        float s   = bn_gamma[tid] * inv;
        sc[tid] = s;
        bi[tid] = bn_beta[tid] - bn_mean[tid] * s;
        pb[tid] = pred_b[tid];
    }
    __syncthreads();

    float macc[HID];
#pragma unroll
    for (int o = 0; o < HID; ++o) macc[o] = 0.f;

    const float* xb = x + (size_t)b * CIN * LIN;
    for (int l = tid; l < LIN; l += 256) {
        float h[HID];
#pragma unroll
        for (int o = 0; o < HID; ++o) h[o] = 0.f;
        for (int c = 0; c < CIN; ++c) {
            float xc = xb[c * LIN + l];
#pragma unroll
            for (int o = 0; o < HID; ++o) h[o] += pw[o * CIN + c] * xc;
        }
#pragma unroll
        for (int o = 0; o < HID; ++o) {
            float y = (h[o] + pb[o]) * sc[o] + bi[o];
            macc[o] += fmaxf(y, 0.f);
        }
    }

#pragma unroll
    for (int o = 0; o < HID; ++o) red[tid * HID + o] = macc[o];
    __syncthreads();

    for (int s = 128; s > 0; s >>= 1) {
        if (tid < s) {
#pragma unroll
            for (int o = 0; o < HID; ++o)
                red[tid * HID + o] += red[(tid + s) * HID + o];
        }
        __syncthreads();
    }

    if (tid == 0) {
        float ssum = lin_b[0];
#pragma unroll
        for (int o = 0; o < HID; ++o)
            ssum += lin_w[o] * (red[o] * (1.0f / (float)LIN));
        float mult = 0.5f + 1.0f / (1.0f + expf(-ssum));
        float ksz  = rintf((float)KSZ0 * mult);
        ksz = fminf(fmaxf(ksz, 2.0f), 3.0f * (float)KSZ0);

        for (int j = 0; j < MAXK; ++j) {
            float t = -1.0f + 2.0f * (float)j / (float)(MAXK - 1);
            float g = t * (ksz / (float)MAXK);
            float p = (g + 1.0f) * 0.5f * (float)(KSZ0 - 1);
            float f = fminf(fmaxf(floorf(p), 0.0f), (float)(KSZ0 - 1));
            int a0 = (int)f;
            int a1 = min(a0 + 1, KSZ0 - 1);
            i0_out[b * MAXK + j] = a0;
            i1_out[b * MAXK + j] = a1;
            w_out [b * MAXK + j] = p - f;
        }
    }
}

// =====================================================================
// Kernel 1.5: hoisted weight interpolation -> bf16 Wbf[b][o][k]
//   one block per (b,o); base_w slab staged in LDS
// =====================================================================
__global__ __launch_bounds__(256)
void dcb_wprep_kernel(const float* __restrict__ base_w,
                      const int*   __restrict__ i0_ws,
                      const int*   __restrict__ i1_ws,
                      const float* __restrict__ w_ws,
                      __bf16* __restrict__ wbf)
{
    __shared__ float bw_s[CIN * KSZ0];   // 8 KB: base_w[o,:,:]
    __shared__ int   li0[MAXK], li1[MAXK];
    __shared__ float lw[MAXK];

    const int bo  = blockIdx.x;          // b*COUT + o
    const int b   = bo >> 8;             // COUT == 256
    const int o   = bo & 255;
    const int tid = threadIdx.x;

    for (int e = tid; e < CIN * KSZ0; e += 256)
        bw_s[e] = base_w[(size_t)o * CIN * KSZ0 + e];
    if (tid < MAXK) {
        li0[tid] = i0_ws[b * MAXK + tid];
        li1[tid] = i1_ws[b * MAXK + tid];
        lw [tid] = w_ws [b * MAXK + tid];
    }
    __syncthreads();

    __bf16* dst = wbf + (size_t)bo * KTOT;
#pragma unroll
    for (int i = 0; i < KTOT / 256; ++i) {
        int k = i * 256 + tid;
        int c = k / MAXK;
        int j = k - c * MAXK;
        float wj = lw[j];
        float v  = bw_s[c * KSZ0 + li0[j]] * (1.0f - wj)
                 + bw_s[c * KSZ0 + li1[j]] * wj;
        dst[k] = (__bf16)v;
    }
}

// =====================================================================
// Kernel 2: per-batch implicit GEMM via bf16 WMMA
//   grid: (17, 2, 32); 256 threads = 8 waves, each owns a 32x64 subtile
//   PRECOMP=true : A tile DMA'd from Wbf by the Tensor Data Mover
//   PRECOMP=false: A tile lerped from base_w in-kernel (ws too small)
// =====================================================================
template <bool PRECOMP>
__global__ __launch_bounds__(256)
void dcb_conv_wmma_kernel(const float* __restrict__ x,
                          const float* __restrict__ base_w,
                          const __bf16* __restrict__ wbf,
                          const int*   __restrict__ i0_ws,
                          const int*   __restrict__ i1_ws,
                          const float* __restrict__ w_ws,
                          float* __restrict__ out)
{
    constexpr int KSTEP = 64;
    constexpr int LDST  = 72;   // padded row stride (elements): 144B rows
    __shared__ __align__(16) __bf16 As[128 * LDST];   // 18 KB  [m][k]
    __shared__ __align__(16) __bf16 Bs[128 * LDST];   // 18 KB  [n][k]
    __shared__ int   li0[MAXK], li1[MAXK];
    __shared__ float lw[MAXK];

    const int tid = threadIdx.x;
    const int bN  = blockIdx.x;
    const int bM  = blockIdx.y;
    const int b   = blockIdx.z;
    const int n0  = bN * 128;
    const int m0  = bM * 128;

    if constexpr (!PRECOMP) {
        if (tid < MAXK) {
            li0[tid] = i0_ws[b * MAXK + tid];
            li1[tid] = i1_ws[b * MAXK + tid];
            lw [tid] = w_ws [b * MAXK + tid];
        }
        __syncthreads();
    }

    const int lane = tid & 31;
    const int wave = tid >> 5;
    const int wm   = wave & 3;   // 0..3 -> 32-row strip
    const int wn   = wave >> 2;  // 0..1 -> 64-col strip

    v8f acc[2][4] = {};

    const float*  xb   = x + (size_t)b * CIN * LIN;
    const __bf16* wrow = PRECOMP ? (wbf + (size_t)(b * COUT + m0) * KTOT) : nullptr;

    // B-fill thread mapping: n fixed per thread -> coalesced x reads
    const int nfix  = tid & 127;
    const int klb   = tid >> 7;            // 0 or 1
    const int lbase = n0 + nfix - PADW;

    for (int kk = 0; kk < KTOT; kk += KSTEP) {
        // ---------------- A tile ----------------
        if constexpr (PRECOMP) {
#if USE_TDM
            if (tid < 32) {
                // Tensor DMA: 2D tile 64 x 128 of bf16, global row stride
                // KTOT elems, LDS padded 4 dwords every 32 dwords (-> 144B rows)
                unsigned long long ga =
                    (unsigned long long)(uintptr_t)(wrow + kk);
                unsigned int lds_lo = (unsigned int)(uintptr_t)(&As[0]);
                uint32x4 g0;
                g0.x = 1u;                               // count=1 (valid)
                g0.y = lds_lo;                           // lds_addr
                g0.z = (unsigned int)ga;                 // global_addr[31:0]
                g0.w = ((unsigned int)(ga >> 32) & 0x1FFFFFFu) | (2u << 30); // addr[56:32] | type=2
                int32x8 g1;
                g1[0] = (1 << 16)      // data_size = 2B
                      | (1 << 20)      // pad_enable
                      | (4 << 22)      // pad_interval: 32 dwords
                      | (3 << 25);     // pad_amount:   4 dwords (16B)
                g1[1] = (KTOT & 0xFFFF) << 16;                    // tensor_dim0 lo16
                g1[2] = ((KTOT >> 16) & 0xFFFF)
                      | ((BATCH * COUT) & 0xFFFF) << 16;          // dim0 hi | dim1 lo
                g1[3] = (((BATCH * COUT) >> 16) & 0xFFFF)
                      | (KSTEP << 16);                            // dim1 hi | tile_dim0=64
                g1[4] = 128;                                      // tile_dim1=128
                g1[5] = KTOT;                                     // tensor_dim0_stride lo32
                g1[6] = 0;
                g1[7] = 0;
                int32x4 g2 = {0, 0, 0, 0};
                int32x4 g3 = {0, 0, 0, 0};
                int32x8 g4 = {0, 0, 0, 0, 0, 0, 0, 0};
                __builtin_amdgcn_tensor_load_to_lds(g0, g1, g2, g3, g4, 0);
            }
#else
            {   // vectorized LDS copy: 2 threads per row, 64B each
                int m   = tid >> 1;
                int kl0 = (tid & 1) * 32;
                const v8bf* s = (const v8bf*)(wrow + (size_t)m * KTOT + kk + kl0);
                v8bf* d = (v8bf*)&As[m * LDST + kl0];
                d[0] = s[0]; d[1] = s[1]; d[2] = s[2]; d[3] = s[3];
            }
#endif
        } else {
            // in-kernel lerp fill: 32 elems/thread, coalesced over k
#pragma unroll
            for (int i = 0; i < 32; ++i) {
                int e  = i * 256 + tid;
                int kl = e & 63;
                int m  = e >> 6;
                int k  = kk + kl;
                int c  = k / MAXK;
                int j  = k - c * MAXK;
                const float* wp = base_w + ((size_t)(m0 + m) * CIN + c) * KSZ0;
                float wj = lw[j];
                float v  = wp[li0[j]] * (1.0f - wj) + wp[li1[j]] * wj;
                As[m * LDST + kl] = (__bf16)v;
            }
        }

        // ---------------- B tile: im2col gather, coalesced over l ----------------
#pragma unroll
        for (int i = 0; i < 32; ++i) {
            int kl = klb + 2 * i;
            int k  = kk + kl;
            int c  = k / MAXK;
            int j  = k - c * MAXK;
            int l  = lbase + j;
            float v = (l >= 0 && l < LIN) ? xb[c * LIN + l] : 0.0f;
            Bs[nfix * LDST + kl] = (__bf16)v;
        }

#if USE_TDM
        if constexpr (PRECOMP) {
            if (tid < 32) wait_tensorcnt0();
        }
#endif
        __syncthreads();

        // ---------------- WMMA: 2 K-chunks of 32 ----------------
        const int khA  = (lane >> 4) * 8;
        const int rowb = wm * 32 + (lane & 15);
        const int kbB  = (lane >> 4) * 8;
        const int colb = wn * 64 + (lane & 15);
#pragma unroll
        for (int ks = 0; ks < KSTEP; ks += 32) {
            v16bf af[2];
#pragma unroll
            for (int mi = 0; mi < 2; ++mi) {
                const __bf16* p = &As[(rowb + mi * 16) * LDST + ks + khA];
                v8bf lo = *(const v8bf*)(p);
                v8bf hi = *(const v8bf*)(p + 16);
                af[mi] = __builtin_shufflevector(lo, hi,
                         0,1,2,3,4,5,6,7,8,9,10,11,12,13,14,15);
            }
#pragma unroll
            for (int ni = 0; ni < 4; ++ni) {
                const __bf16* p = &Bs[(colb + ni * 16) * LDST + ks + kbB];
                v8bf lo = *(const v8bf*)(p);
                v8bf hi = *(const v8bf*)(p + 16);
                v16bf bf = __builtin_shufflevector(lo, hi,
                           0,1,2,3,4,5,6,7,8,9,10,11,12,13,14,15);
#pragma unroll
                for (int mi = 0; mi < 2; ++mi) {
                    acc[mi][ni] = __builtin_amdgcn_wmma_f32_16x16x32_bf16(
                        false, af[mi], false, bf,
                        (short)0, acc[mi][ni], false, false);
                }
            }
        }
        __syncthreads();
    }

    // ---------------- epilogue: 16x16 f32 C/D layout ----------------
    const int madd = (lane >> 4) * 8;
    const int ncol = lane & 15;
#pragma unroll
    for (int mi = 0; mi < 2; ++mi) {
#pragma unroll
        for (int ni = 0; ni < 4; ++ni) {
#pragma unroll
            for (int r = 0; r < 8; ++r) {
                int row = m0 + wm * 32 + mi * 16 + r + madd;
                int col = n0 + wn * 64 + ni * 16 + ncol;
                if (col < LOUT) {
                    out[((size_t)b * COUT + row) * LOUT + col] = acc[mi][ni][r];
                }
            }
        }
    }
}

// =====================================================================
extern "C" void kernel_launch(void* const* d_in, const int* in_sizes, int n_in,
                              void* d_out, int out_size, void* d_ws, size_t ws_size,
                              hipStream_t stream)
{
    const float* x        = (const float*)d_in[0];
    const float* base_w   = (const float*)d_in[1];
    const float* pred_w   = (const float*)d_in[2];
    const float* pred_b   = (const float*)d_in[3];
    const float* bn_gamma = (const float*)d_in[4];
    const float* bn_beta  = (const float*)d_in[5];
    const float* bn_mean  = (const float*)d_in[6];
    const float* bn_var   = (const float*)d_in[7];
    const float* lin_w    = (const float*)d_in[8];
    const float* lin_b    = (const float*)d_in[9];
    float* out = (float*)d_out;

    // ws layout: [i0: B*MAXK ints][i1: B*MAXK ints][w: B*MAXK floats] ... [Wbf @16KB]
    int*   i0_ws = (int*)d_ws;
    int*   i1_ws = i0_ws + BATCH * MAXK;
    float* w_ws  = (float*)(i1_ws + BATCH * MAXK);
    const size_t wbf_off   = 16384;
    const size_t wbf_bytes = (size_t)BATCH * COUT * KTOT * sizeof(__bf16);
    __bf16* wbf = (__bf16*)((char*)d_ws + wbf_off);
    const bool precomp = (ws_size >= wbf_off + wbf_bytes);

    dcb_predict_kernel<<<BATCH, 256, 0, stream>>>(
        x, pred_w, pred_b, bn_gamma, bn_beta, bn_mean, bn_var,
        lin_w, lin_b, i0_ws, i1_ws, w_ws);

    dim3 grid((LOUT + 127) / 128, COUT / 128, BATCH);
    if (precomp) {
        dcb_wprep_kernel<<<BATCH * COUT, 256, 0, stream>>>(
            base_w, i0_ws, i1_ws, w_ws, wbf);
        dcb_conv_wmma_kernel<true><<<grid, 256, 0, stream>>>(
            x, base_w, wbf, i0_ws, i1_ws, w_ws, out);
    } else {
        dcb_conv_wmma_kernel<false><<<grid, 256, 0, stream>>>(
            x, base_w, nullptr, i0_ws, i1_ws, w_ws, out);
    }
}
